// PELTRA_17952963297345
// MI455X (gfx1250) — matrix operand order repaired and checked by reading, not compiled, not run on hardware
//
#include <hip/hip_runtime.h>
#include <hip/hip_bf16.h>
#include <math.h>

typedef __attribute__((ext_vector_type(16))) _Float16 v16h;
typedef __attribute__((ext_vector_type(8)))  float    v8f;

#define SEQ   2048
#define DM    34
#define NH    2
#define HD    17
#define NL    8
#define DFF   2048
#define DPAD  64     // padded K for D-dim GEMMs (2 WMMA k-steps)
#define QKVP  112    // 3*D=102 padded to 7 n-tiles
#define DPO   48     // D=34 padded to 3 n-tiles
#define HDP   32     // head dim 17 padded to 32

// ---------------------------------------------------------------------------
// Weight pack: fp32 [N][K] -> fp16 [Npad][Kpad], zero pad.
// ---------------------------------------------------------------------------
__global__ void pack_w(const float* __restrict__ src, _Float16* __restrict__ dst,
                       int N, int K, int Npad, int Kpad) {
  int idx = blockIdx.x * blockDim.x + threadIdx.x;
  if (idx >= Npad * Kpad) return;
  int n = idx / Kpad, k = idx - n * Kpad;
  float v = (n < N && k < K) ? src[n * K + k] : 0.0f;
  dst[idx] = (_Float16)v;
}

// ---------------------------------------------------------------------------
// Embedding + sinusoidal PE. Writes fp32 master h and fp16 padded mirror.
// ---------------------------------------------------------------------------
__global__ void embed_pe(const float* __restrict__ x, float* __restrict__ h32,
                         _Float16* __restrict__ h16) {
  int s = blockIdx.x * blockDim.x + threadIdx.x;
  if (s >= SEQ) return;
  const float c = -logf(10000.0f) / (float)DM;
  #pragma unroll 1
  for (int i = 0; i < HD; ++i) {
    float div = __expf((2.0f * i) * c);
    float ang = (float)s * div;
    float v0 = x[s * DM + 2 * i]     + __sinf(ang);
    float v1 = x[s * DM + 2 * i + 1] + __cosf(ang);
    h32[s * DM + 2 * i]     = v0;
    h32[s * DM + 2 * i + 1] = v1;
    h16[s * DPAD + 2 * i]     = (_Float16)v0;
    h16[s * DPAD + 2 * i + 1] = (_Float16)v1;
  }
  for (int d = DM; d < DPAD; ++d) h16[s * DPAD + d] = (_Float16)0.0f;
}

// ---------------------------------------------------------------------------
// Wave-level WMMA GEMM body, tile count specialized at compile time so the
// k-loop is branch-free: 1 A-operand clause + NT B clauses + NT v_wmma per
// k-step, A registers reused across all NT tiles.
// ---------------------------------------------------------------------------
template <int NT>
__device__ __forceinline__
void gemm_wave(const _Float16* __restrict__ arow,
               const _Float16* __restrict__ B, int ldb, int Ksteps,
               const float* __restrict__ bias, int Nreal,
               float* __restrict__ Cf32, _Float16* __restrict__ Cf16,
               int ldc, int relu, int mt, int ntb, int l15, int kh) {
  v8f acc[NT] = {};
  for (int ks = 0; ks < Ksteps; ++ks) {
    const int kb = ks * 32;
    v16h a;
    #pragma unroll
    for (int j = 0; j < 8; ++j) {       // A: contiguous k = 16*kh + s (same order as B)
      int k = 2 * j + 16 * kh;
      a[2 * j]     = arow[kb + k];
      a[2 * j + 1] = arow[kb + k + 1];
    }
    #pragma unroll
    for (int i = 0; i < NT; ++i) {
      const _Float16* brow = B + ((ntb + i) * 16 + l15) * ldb + kb;
      v16h b;
      #pragma unroll
      for (int j = 0; j < 8; ++j) {     // B: contiguous k = 16*kh + s (same order as A)
        int k = 2 * j + 16 * kh;
        b[2 * j]     = brow[k];
        b[2 * j + 1] = brow[k + 1];
      }
      acc[i] = __builtin_amdgcn_wmma_f32_16x16x32_f16(false, a, false, b,
                                                      (short)0, acc[i], false, false);
    }
  }
  #pragma unroll
  for (int i = 0; i < NT; ++i) {
    const int col = (ntb + i) * 16 + l15;
    const float bv = (bias != nullptr && col < Nreal) ? bias[col] : 0.0f;
    #pragma unroll
    for (int r = 0; r < 8; ++r) {
      int row = mt * 16 + r + 8 * kh;
      float v = acc[i][r] + bv;
      if (relu) v = v > 0.0f ? v : 0.0f;
      if (Cf32) Cf32[row * ldc + col] = v;
      if (Cf16) Cf16[row * ldc + col] = (_Float16)v;
    }
  }
}

// ---------------------------------------------------------------------------
// WMMA GEMM: C[M][N] = A[M][K] * B[N][K]^T + bias.
// Block = 256 threads = 8 waves; wave w owns M-tile (blockIdx.y*8+w).
// Each wave computes up to 4 consecutive N-tiles; the remainder count is
// resolved once (wave-uniform switch), keeping EXEC all-1s for every WMMA.
// grid = (ceil(Ntiles/4), Mtiles/8). M/N lane mapping per CDNA5 ISA 7.12.2; k order contiguous.
// ---------------------------------------------------------------------------
__global__ __launch_bounds__(256)
void gemm_wmma(const _Float16* __restrict__ A, int lda, int Ksteps,
               const _Float16* __restrict__ B, int ldb,
               const float* __restrict__ bias, int Nreal, int Ntiles,
               float* __restrict__ Cf32, _Float16* __restrict__ Cf16,
               int ldc, int relu) {
  const int wave = threadIdx.x >> 5;
  const int lane = threadIdx.x & 31;
  const int l15  = lane & 15;
  const int kh   = lane >> 4;
  const int mt   = blockIdx.y * 8 + wave;
  const int ntb  = blockIdx.x * 4;

  const _Float16* arow = A + (mt * 16 + l15) * lda;
  const int nrem = Ntiles - ntb;
  if (nrem >= 4) {
    gemm_wave<4>(arow, B, ldb, Ksteps, bias, Nreal, Cf32, Cf16, ldc, relu, mt, ntb, l15, kh);
  } else if (nrem == 3) {
    gemm_wave<3>(arow, B, ldb, Ksteps, bias, Nreal, Cf32, Cf16, ldc, relu, mt, ntb, l15, kh);
  } else if (nrem == 2) {
    gemm_wave<2>(arow, B, ldb, Ksteps, bias, Nreal, Cf32, Cf16, ldc, relu, mt, ntb, l15, kh);
  } else if (nrem == 1) {
    gemm_wave<1>(arow, B, ldb, Ksteps, bias, Nreal, Cf32, Cf16, ldc, relu, mt, ntb, l15, kh);
  }
}

// ---------------------------------------------------------------------------
// Split qkv fp32 [S][112] into per-head padded fp16 Q (pre-scaled), K, V
// buffers [head][S][32]; zero the attention-output staging row pads.
// ---------------------------------------------------------------------------
__global__ void split_qkv(const float* __restrict__ qkv,
                          _Float16* __restrict__ Q, _Float16* __restrict__ K,
                          _Float16* __restrict__ V, _Float16* __restrict__ attO) {
  int s = blockIdx.x * blockDim.x + threadIdx.x;
  if (s >= SEQ) return;
  const float scale = 1.0f / sqrtf((float)HD);
  for (int h = 0; h < NH; ++h) {
    for (int d = 0; d < HDP; ++d) {
      float q = 0.f, k = 0.f, v = 0.f;
      if (d < HD) {
        q = qkv[s * QKVP + h * HD + d] * scale;
        k = qkv[s * QKVP + DM + h * HD + d];
        v = qkv[s * QKVP + 2 * DM + h * HD + d];
      }
      int o = (h * SEQ + s) * HDP + d;
      Q[o] = (_Float16)q; K[o] = (_Float16)k; V[o] = (_Float16)v;
    }
  }
  for (int d = 0; d < DPAD; ++d) attO[s * DPAD + d] = (_Float16)0.0f;
}

// ---------------------------------------------------------------------------
// Flash attention. Block = 256 threads = 8 waves; wave w owns q-tile
// (blockIdx.x*8+w) of 16 rows for head blockIdx.y. Key blocks of 32.
// Scores: 2x v_wmma. Online softmax with __shfl_xor 16-lane row reductions.
// P re-layout C->A via per-wave LDS slice (2KB each). PV: 2x v_wmma.
// All waves run the same trip count, so the block-wide barriers are legal.
// ---------------------------------------------------------------------------
__global__ __launch_bounds__(256)
void attn_flash(const _Float16* __restrict__ Q, const _Float16* __restrict__ K,
                const _Float16* __restrict__ V, _Float16* __restrict__ O) {
  const int wave = threadIdx.x >> 5;
  const int lane = threadIdx.x & 31;
  const int l15 = lane & 15, kh = lane >> 4;
  const int qt = blockIdx.x * 8 + wave;
  const int head = blockIdx.y;

  const _Float16* Qh = Q + (size_t)head * SEQ * HDP;
  const _Float16* Kh = K + (size_t)head * SEQ * HDP;
  const _Float16* Vh = V + (size_t)head * SEQ * HDP;

  // Q tile as A operand (16x32)
  v16h aq;
  {
    const _Float16* qrow = Qh + (qt * 16 + l15) * HDP;
    #pragma unroll
    for (int j = 0; j < 8; ++j) {
      int k = 2 * j + 16 * kh;
      aq[2 * j] = qrow[k]; aq[2 * j + 1] = qrow[k + 1];
    }
  }

  v8f o0 = {}, o1 = {};
  float m[8], lsum[8];
  #pragma unroll
  for (int r = 0; r < 8; ++r) { m[r] = -1e30f; lsum[r] = 0.0f; }

  __shared__ float ptile_all[8 * 16 * 32];
  float* ptile = ptile_all + wave * (16 * 32);

  for (int t = 0; t < SEQ; t += 32) {
    // K rows as B operands: 16 keys each
    v16h bk0, bk1;
    {
      const _Float16* k0 = Kh + (t + l15) * HDP;
      const _Float16* k1 = Kh + (t + 16 + l15) * HDP;
      #pragma unroll
      for (int j = 0; j < 8; ++j) {
        int k = 2 * j + 16 * kh;
        bk0[2 * j] = k0[k]; bk0[2 * j + 1] = k0[k + 1];
        bk1[2 * j] = k1[k]; bk1[2 * j + 1] = k1[k + 1];
      }
    }
    v8f z = {};
    v8f s0 = __builtin_amdgcn_wmma_f32_16x16x32_f16(false, aq, false, bk0,
                                                    (short)0, z, false, false);
    v8f s1 = __builtin_amdgcn_wmma_f32_16x16x32_f16(false, aq, false, bk1,
                                                    (short)0, z, false, false);
    // online softmax per row (row = r + 8*kh, cols across 16-lane group)
    #pragma unroll
    for (int r = 0; r < 8; ++r) {
      float mx = fmaxf(s0[r], s1[r]);
      #pragma unroll
      for (int off = 1; off < 16; off <<= 1) mx = fmaxf(mx, __shfl_xor(mx, off, 32));
      float mnew = fmaxf(m[r], mx);
      float corr = __expf(m[r] - mnew);
      m[r] = mnew;
      float p0 = __expf(s0[r] - mnew);
      float p1 = __expf(s1[r] - mnew);
      float ps = p0 + p1;
      #pragma unroll
      for (int off = 1; off < 16; off <<= 1) ps += __shfl_xor(ps, off, 32);
      lsum[r] = lsum[r] * corr + ps;
      o0[r] *= corr; o1[r] *= corr;
      int row = r + 8 * kh;
      ptile[row * 32 + l15]      = p0;
      ptile[row * 32 + 16 + l15] = p1;
    }
    __syncthreads();
    // P tile back as A operand (16 rows x 32 keys), fp16
    v16h ap;
    #pragma unroll
    for (int j = 0; j < 8; ++j) {
      int k = 2 * j + 16 * kh;
      ap[2 * j]     = (_Float16)ptile[l15 * 32 + k];
      ap[2 * j + 1] = (_Float16)ptile[l15 * 32 + k + 1];
    }
    __syncthreads();
    // V block as B operands: K-dim = 32 keys, N = vdims 0-15 / 16-31
    v16h bv0, bv1;
    #pragma unroll
    for (int j = 0; j < 8; ++j) {
      int key = t + 2 * j + 16 * kh;
      bv0[2 * j]     = Vh[key * HDP + l15];
      bv0[2 * j + 1] = Vh[(key + 1) * HDP + l15];
      bv1[2 * j]     = Vh[key * HDP + 16 + l15];
      bv1[2 * j + 1] = Vh[(key + 1) * HDP + 16 + l15];
    }
    o0 = __builtin_amdgcn_wmma_f32_16x16x32_f16(false, ap, false, bv0,
                                                (short)0, o0, false, false);
    o1 = __builtin_amdgcn_wmma_f32_16x16x32_f16(false, ap, false, bv1,
                                                (short)0, o1, false, false);
  }

  #pragma unroll
  for (int r = 0; r < 8; ++r) {
    int row = qt * 16 + r + 8 * kh;
    float inv = 1.0f / lsum[r];
    int vd0 = l15, vd1 = 16 + l15;
    if (vd0 < HD) O[row * DPAD + head * HD + vd0] = (_Float16)(o0[r] * inv);
    if (vd1 < HD) O[row * DPAD + head * HD + vd1] = (_Float16)(o1[r] * inv);
  }
}

// ---------------------------------------------------------------------------
// Residual add + layernorm. Updates fp32 master h and fp16 padded mirror.
// ---------------------------------------------------------------------------
__global__ void resid_ln(float* __restrict__ h32, _Float16* __restrict__ h16,
                         const float* __restrict__ c, int ldc,
                         const float* __restrict__ g, const float* __restrict__ b) {
  int s = blockIdx.x * blockDim.x + threadIdx.x;
  if (s >= SEQ) return;
  float v[DM];
  float mean = 0.0f;
  for (int d = 0; d < DM; ++d) { v[d] = h32[s * DM + d] + c[s * ldc + d]; mean += v[d]; }
  mean *= (1.0f / DM);
  float var = 0.0f;
  for (int d = 0; d < DM; ++d) { float t = v[d] - mean; var += t * t; }
  var *= (1.0f / DM);
  float rstd = rsqrtf(var + 1e-5f);
  for (int d = 0; d < DM; ++d) {
    float o = g[d] * (v[d] - mean) * rstd + b[d];
    h32[s * DM + d] = o;
    h16[s * DPAD + d] = (_Float16)o;
  }
  for (int d = DM; d < DPAD; ++d) h16[s * DPAD + d] = (_Float16)0.0f;
}

// ---------------------------------------------------------------------------
// Final head: tanh(h @ lout^T + b) then Gram-Schmidt rotation assembly.
// ---------------------------------------------------------------------------
__global__ void head_out(const float* __restrict__ h32,
                         const float* __restrict__ lw, const float* __restrict__ lb,
                         float* __restrict__ out) {
  int s = blockIdx.x * blockDim.x + threadIdx.x;
  if (s >= SEQ) return;
  float y[96];
  for (int n = 0; n < 96; ++n) {
    float acc = lb[n];
    for (int k = 0; k < DM; ++k) acc += h32[s * DM + k] * lw[n * DM + k];
    y[n] = tanhf(acc);
  }
  for (int gi = 0; gi < 16; ++gi) {
    float a1[3] = { y[gi * 6 + 0], y[gi * 6 + 1], y[gi * 6 + 2] };
    float a2[3] = { y[gi * 6 + 3], y[gi * 6 + 4], y[gi * 6 + 5] };
    float n1 = rsqrtf(a1[0]*a1[0] + a1[1]*a1[1] + a1[2]*a1[2]);
    float r1[3] = { a1[0]*n1, a1[1]*n1, a1[2]*n1 };
    float dp = r1[0]*a2[0] + r1[1]*a2[1] + r1[2]*a2[2];
    float t2[3] = { a2[0]-dp*r1[0], a2[1]-dp*r1[1], a2[2]-dp*r1[2] };
    float n2 = rsqrtf(t2[0]*t2[0] + t2[1]*t2[1] + t2[2]*t2[2]);
    float r2[3] = { t2[0]*n2, t2[1]*n2, t2[2]*n2 };
    float c3[3] = { r1[1]*r2[2]-r1[2]*r2[1], r1[2]*r2[0]-r1[0]*r2[2], r1[0]*r2[1]-r1[1]*r2[0] };
    float n3 = rsqrtf(c3[0]*c3[0] + c3[1]*c3[1] + c3[2]*c3[2]);
    float r3[3] = { c3[0]*n3, c3[1]*n3, c3[2]*n3 };
    float* o = out + ((size_t)s * 16 + gi) * 9;
    o[0]=r1[0]; o[1]=r2[0]; o[2]=r3[0];
    o[3]=r1[1]; o[4]=r2[1]; o[5]=r3[1];
    o[6]=r1[2]; o[7]=r2[2]; o[8]=r3[2];
  }
}

// ---------------------------------------------------------------------------
extern "C" void kernel_launch(void* const* d_in, const int* in_sizes, int n_in,
                              void* d_out, int out_size, void* d_ws, size_t ws_size,
                              hipStream_t stream) {
  (void)in_sizes; (void)n_in; (void)out_size; (void)ws_size;
  const float* x          = (const float*)d_in[0];
  const float* in_proj_w  = (const float*)d_in[1];
  const float* in_proj_b  = (const float*)d_in[2];
  const float* out_proj_w = (const float*)d_in[3];
  const float* out_proj_b = (const float*)d_in[4];
  const float* ln1_g      = (const float*)d_in[5];
  const float* ln1_b      = (const float*)d_in[6];
  const float* lin1_w     = (const float*)d_in[7];
  const float* lin1_b     = (const float*)d_in[8];
  const float* lin2_w     = (const float*)d_in[9];
  const float* lin2_b     = (const float*)d_in[10];
  const float* ln2_g      = (const float*)d_in[11];
  const float* ln2_b      = (const float*)d_in[12];
  const float* lout_w     = (const float*)d_in[13];
  const float* lout_b     = (const float*)d_in[14];
  float* out = (float*)d_out;

  // bump allocator over d_ws
  char* base = (char*)d_ws;
  size_t off = 0;
  auto alloc = [&](size_t bytes) -> char* {
    char* p = base + off;
    off = (off + bytes + 255) & ~(size_t)255;
    return p;
  };
  _Float16* wInproj  = (_Float16*)alloc((size_t)NL * QKVP * DPAD * 2);
  _Float16* wOutproj = (_Float16*)alloc((size_t)NL * DPO * DPAD * 2);
  _Float16* wLin1    = (_Float16*)alloc((size_t)NL * DFF * DPAD * 2);
  _Float16* wLin2    = (_Float16*)alloc((size_t)NL * DPO * DFF * 2);
  float*    h32      = (float*)   alloc((size_t)SEQ * DM * 4);
  _Float16* h16      = (_Float16*)alloc((size_t)SEQ * DPAD * 2);
  float*    qkv32    = (float*)   alloc((size_t)SEQ * QKVP * 4);
  _Float16* qf       = (_Float16*)alloc((size_t)NH * SEQ * HDP * 2);
  _Float16* kf       = (_Float16*)alloc((size_t)NH * SEQ * HDP * 2);
  _Float16* vf       = (_Float16*)alloc((size_t)NH * SEQ * HDP * 2);
  _Float16* atto16   = (_Float16*)alloc((size_t)SEQ * DPAD * 2);
  float*    c32      = (float*)   alloc((size_t)SEQ * DPO * 4);
  _Float16* ff16     = (_Float16*)alloc((size_t)SEQ * DFF * 2);

  // ---- pack weights (fp32 -> padded fp16) ----
  for (int l = 0; l < NL; ++l) {
    pack_w<<<(QKVP * DPAD + 255) / 256, 256, 0, stream>>>(
        in_proj_w + (size_t)l * 3 * DM * DM, wInproj + (size_t)l * QKVP * DPAD,
        3 * DM, DM, QKVP, DPAD);
    pack_w<<<(DPO * DPAD + 255) / 256, 256, 0, stream>>>(
        out_proj_w + (size_t)l * DM * DM, wOutproj + (size_t)l * DPO * DPAD,
        DM, DM, DPO, DPAD);
    pack_w<<<(DFF * DPAD + 255) / 256, 256, 0, stream>>>(
        lin1_w + (size_t)l * DFF * DM, wLin1 + (size_t)l * DFF * DPAD,
        DFF, DM, DFF, DPAD);
    pack_w<<<(DPO * DFF + 255) / 256, 256, 0, stream>>>(
        lin2_w + (size_t)l * DM * DFF, wLin2 + (size_t)l * DPO * DFF,
        DM, DFF, DPO, DFF);
  }

  // ---- embed + PE ----
  embed_pe<<<SEQ / 256, 256, 0, stream>>>(x, h32, h16);

  const int MB = SEQ / 16 / 8; // 16 blocks of 8 M-tile waves
  for (int l = 0; l < NL; ++l) {
    // qkv = h @ Wqkv^T + b   (M=2048, N=102->112, K=64)
    gemm_wmma<<<dim3((QKVP / 16 + 3) / 4, MB), 256, 0, stream>>>(
        h16, DPAD, 2, wInproj + (size_t)l * QKVP * DPAD, DPAD,
        in_proj_b + (size_t)l * 3 * DM, 3 * DM, QKVP / 16, qkv32, nullptr, QKVP, 0);
    split_qkv<<<SEQ / 256, 256, 0, stream>>>(qkv32, qf, kf, vf, atto16);
    // flash attention: 8 q-tile waves per block, one head per blockIdx.y
    attn_flash<<<dim3(SEQ / 16 / 8, NH), 256, 0, stream>>>(qf, kf, vf, atto16);
    // out-proj (M=2048, N=34->48, K=64)
    gemm_wmma<<<dim3(1, MB), 256, 0, stream>>>(
        atto16, DPAD, 2, wOutproj + (size_t)l * DPO * DPAD, DPAD,
        out_proj_b + (size_t)l * DM, DM, DPO / 16, c32, nullptr, DPO, 0);
    resid_ln<<<SEQ / 256, 256, 0, stream>>>(h32, h16, c32, DPO,
                                            ln1_g + (size_t)l * DM, ln1_b + (size_t)l * DM);
    // FFN1 + ReLU -> fp16 (M=2048, N=2048, K=64)
    gemm_wmma<<<dim3(DFF / 16 / 4, MB), 256, 0, stream>>>(
        h16, DPAD, 2, wLin1 + (size_t)l * DFF * DPAD, DPAD,
        lin1_b + (size_t)l * DFF, DFF, DFF / 16, nullptr, ff16, DFF, 1);
    // FFN2 (M=2048, N=34->48, K=2048)
    gemm_wmma<<<dim3(1, MB), 256, 0, stream>>>(
        ff16, DFF, DFF / 32, wLin2 + (size_t)l * DPO * DFF, DFF,
        lin2_b + (size_t)l * DM, DM, DPO / 16, c32, nullptr, DPO, 0);
    resid_ln<<<SEQ / 256, 256, 0, stream>>>(h32, h16, c32, DPO,
                                            ln2_g + (size_t)l * DM, ln2_b + (size_t)l * DM);
  }

  // ---- final head + Gram-Schmidt ----
  head_out<<<SEQ / 256, 256, 0, stream>>>(h32, lout_w, lout_b, out);
}
